// RNN_NET_77481210020335
// MI455X (gfx1250) — matrix operand order repaired and checked by reading, not compiled, not run on hardware
//
#include <hip/hip_runtime.h>
#include <cstdint>
#include <cstddef>

// ---------------- problem constants ----------------
#define B_    70
#define T_    1000
#define I_    20
#define H_    128          // hidden
#define G_    384          // 3*H
#define ROWS_ 80           // B padded to 5 x 16 WMMA row tiles

typedef __attribute__((ext_vector_type(16))) _Float16 v16h;
typedef __attribute__((ext_vector_type(8)))  float    v8f;
typedef __attribute__((ext_vector_type(4)))  int      v4i;

union AFrag { v16h v; int4 q[2]; };

static __device__ __forceinline__ v8f wmma_f16(v16h a, v16h b, v8f c) {
  // D = A(16x32 f16) * B(32x16 f16) + C(16x16 f32)
  return __builtin_amdgcn_wmma_f32_16x16x32_f16(false, a, false, b,
                                                (short)0, c, false, false);
}

static __device__ __forceinline__ float sigmoidf_(float x) {
  return 1.0f / (1.0f + __expf(-x));
}
static __device__ __forceinline__ float tanhf_(float x) {
  float e = __expf(2.0f * x);
  return 1.0f - 2.0f / (e + 1.0f);
}

// ---------------- async global->LDS copy (CDNA5 path) ----------------
#if __has_builtin(__builtin_amdgcn_global_load_async_to_lds_b128)
#define HAVE_ASYNC_LDS 1
#else
#define HAVE_ASYNC_LDS 0
#endif

// builtin signature (from hipcc diagnostic): param1 = v4i addrspace(1)*,
// param2 = LDS-side pointer; addrspace(1) prints as __device__ in HIP mode.
#define GPTR1(p) ((__attribute__((address_space(1))) v4i*)(unsigned long long)(uintptr_t)(p))
#define LPTR3(p) ((__attribute__((address_space(3))) v4i*)(unsigned int)(uintptr_t)(p))

// Copies one timestep of gx (B_*G_ = 26880 floats = 6720 x b128) into LDS.
// Exactly 27 async instructions per wave -> uniform ASYNCcnt bookkeeping.
static __device__ __forceinline__ void gx_fetch(const float* __restrict__ src,
                                                float* dst_lds, int tid) {
  const float4* s4 = (const float4*)src;
  float4*       d4 = (float4*)dst_lds;
#pragma unroll
  for (int i = 0; i < 27; ++i) {
    int idx = tid + (i << 8);
    if (idx > 6719) idx = 6719;     // clamp: duplicate transfer, keeps count uniform
#if HAVE_ASYNC_LDS
    __builtin_amdgcn_global_load_async_to_lds_b128(GPTR1(s4 + idx), LPTR3(d4 + idx), 0, 0);
#else
    d4[idx] = s4[idx];
#endif
  }
}

static __device__ __forceinline__ void gx_wait() {
#if HAVE_ASYNC_LDS
#if __has_builtin(__builtin_amdgcn_s_wait_asynccnt)
  __builtin_amdgcn_s_wait_asynccnt(27);   // leave only the just-issued next-step batch
#else
  asm volatile("s_wait_asynccnt 27" ::: "memory");
#endif
#endif
}

// ---------------- kernel 1: weight prep (f32 -> f16, bias folding) ----------------
__global__ __launch_bounds__(256) void prep_kernel(
    const float* __restrict__ whh0f, const float* __restrict__ whh0b,
    const float* __restrict__ whh1f, const float* __restrict__ whh1b,
    const float* __restrict__ wih1f, const float* __restrict__ wih1b,
    const float* __restrict__ bi0f,  const float* __restrict__ bh0f,
    const float* __restrict__ bi0b,  const float* __restrict__ bh0b,
    const float* __restrict__ bi1f,  const float* __restrict__ bh1f,
    const float* __restrict__ bi1b,  const float* __restrict__ bh1b,
    _Float16* __restrict__ whh16, _Float16* __restrict__ wih16,
    float* __restrict__ bsum0, float* __restrict__ bsum1, float* __restrict__ bhhn) {
  const int NW = G_ * H_;              // 49152 per (layer,dir)
  const int NI = G_ * 2 * H_;          // 98304 per dir (layer1 w_ih)
  int idx = blockIdx.x * 256 + threadIdx.x;
  if (idx < 4 * NW) {
    int which = idx / NW, off = idx % NW;
    const float* s = (which == 0) ? whh0f : (which == 1) ? whh0b
                   : (which == 2) ? whh1f : whh1b;
    whh16[idx] = (_Float16)s[off];
  } else if (idx < 4 * NW + 2 * NI) {
    int i2 = idx - 4 * NW;
    int which = i2 / NI, off = i2 % NI;
    wih16[i2] = (_Float16)((which ? wih1b : wih1f)[off]);
  } else if (idx < 4 * NW + 2 * NI + 2 * G_) {
    int i3 = idx - (4 * NW + 2 * NI);
    int dir = i3 / G_, g = i3 % G_;
    const float* bi = dir ? bi0b : bi0f;
    const float* bh = dir ? bh0b : bh0f;
    // fold b_hh only for r/z sections; n-section b_hh stays inside r* term
    bsum0[i3] = bi[g] + (g < 2 * H_ ? bh[g] : 0.0f);
  } else if (idx < 4 * NW + 2 * NI + 4 * G_) {
    int i4 = idx - (4 * NW + 2 * NI + 2 * G_);
    int dir = i4 / G_, g = i4 % G_;
    const float* bi = dir ? bi1b : bi1f;
    const float* bh = dir ? bh1b : bh1f;
    bsum1[i4] = bi[g] + (g < 2 * H_ ? bh[g] : 0.0f);
  } else if (idx < 4 * NW + 2 * NI + 4 * G_ + 4 * H_) {
    int i5 = idx - (4 * NW + 2 * NI + 4 * G_);
    int ld = i5 / H_, jj = i5 % H_;     // ld: l0f,l0b,l1f,l1b
    const float* bh = (ld == 0) ? bh0f : (ld == 1) ? bh0b : (ld == 2) ? bh1f : bh1b;
    bhhn[i5] = bh[2 * H_ + jj];
  }
}

// ---------------- kernel 2: layer-0 input gates (K=20, VALU) ----------------
__global__ __launch_bounds__(256) void gx0_kernel(
    const float* __restrict__ x,       // [B][T][I]
    const float* __restrict__ wf,      // [G][I]
    const float* __restrict__ wb,      // [G][I]
    const float* __restrict__ bsum0,   // [2][G]
    float* __restrict__ gxp0) {        // [2][T][B][G]
  int idx = blockIdx.x * 256 + threadIdx.x;
  if (idx >= 2 * T_ * B_ * G_) return;
  int g   = idx % G_;
  int b   = (idx / G_) % B_;
  int t   = (idx / (G_ * B_)) % T_;
  int dir = idx / (G_ * B_ * T_);
  const float* w  = (dir ? wb : wf) + g * I_;
  const float* xr = x + (size_t)(b * T_ + t) * I_;
  float s = bsum0[dir * G_ + g];
#pragma unroll
  for (int i = 0; i < I_; ++i) s += xr[i] * w[i];
  gxp0[(size_t)dir * T_ * B_ * G_ + (size_t)(t * B_ + b) * G_ + g] = s;
}

// ---------------- kernel 3: layer-1 input gates, WMMA GEMM (K=256) ----------------
// rows = t*B+b (70000 = 4375 tiles of 16 exactly), cols = 384 (24 tiles)
__global__ __launch_bounds__(256) void gx1_gemm_kernel(
    const _Float16* __restrict__ a16,    // [T*B][2H] layer-0 output (f16)
    const _Float16* __restrict__ wih16,  // [2][G][2H]
    const float* __restrict__ bsum1,     // [2][G]
    float* __restrict__ gxp1) {          // [2][T*B][G]
  const int rt  = blockIdx.x;
  const int dir = blockIdx.y;
  const int tid = threadIdx.x;
  const int lane = tid & 31;
  const int wv   = tid >> 5;            // 8 waves: col tiles {wv, wv+8, wv+16}
  const int hi   = (lane >> 4) & 1;
  const int l16  = lane & 15;

  const _Float16* wbp = wih16 + (size_t)dir * G_ * (2 * H_);
  v8f acc[3] = {};
#pragma unroll
  for (int kt = 0; kt < 8; ++kt) {      // K = 256 = 8 x 32
    AFrag a;
    const int row = rt * 16 + l16;
    const int k0  = kt * 32 + hi * 16;
    a.q[0] = *(const int4*)(a16 + (size_t)row * (2 * H_) + k0);
    a.q[1] = *(const int4*)(a16 + (size_t)row * (2 * H_) + k0 + 8);
#pragma unroll
    for (int c = 0; c < 3; ++c) {
      int n = (wv + 8 * c) * 16 + l16;
      v16h bf = *(const v16h*)(wbp + (size_t)n * (2 * H_) + kt * 32 + hi * 16);
      acc[c] = wmma_f16(a.v, bf, acc[c]);
    }
  }
  float* gout = gxp1 + (size_t)dir * T_ * B_ * G_;
#pragma unroll
  for (int c = 0; c < 3; ++c) {
    int g = (wv + 8 * c) * 16 + l16;
    float bias = bsum1[dir * G_ + g];
#pragma unroll
    for (int r = 0; r < 8; ++r) {
      int m = rt * 16 + r + hi * 8;     // D layout: M = vgpr + 8*(lane>=16)
      gout[(size_t)m * G_ + g] = acc[c][r] + bias;
    }
  }
}

// ---------------- kernel 4: the sequential GRU scan (WMMA + async LDS) ----------------
// grid.x = 2 (dir), 256 threads = 8 waves. Wave w owns gate columns j=w*16+l16
// for all three gates (col tiles {w, w+8, w+16}) -> gate math stays in-wave.
template <int LAYER>
__global__ __launch_bounds__(256) void gru_scan_kernel(
    const float* __restrict__ gx_all,    // [2][T][B][G] (biases folded, see prep)
    const _Float16* __restrict__ whh16,  // [2][G][H] for this layer
    const float* __restrict__ h0_all,    // [4][B][H]
    const float* __restrict__ bhhn,      // [2][H] this layer (n-gate b_hh)
    _Float16* __restrict__ out0h,        // [T*B][2H] f16 (layer0 out)
    float* __restrict__ dout) {          // [B][T][2H] f32 (final out)
  extern __shared__ char smem[];
  _Float16* hbuf = (_Float16*)smem;                                    // 2 x 80x128 f16
  float*    gxb  = (float*)(smem + 2 * ROWS_ * H_ * sizeof(_Float16)); // 2 x 26880 f32

  const int dir  = blockIdx.x;
  const int tid  = threadIdx.x;
  const int lane = tid & 31;
  const int wv   = tid >> 5;
  const int hi   = (lane >> 4) & 1;
  const int l16  = lane & 15;
  const int j    = wv * 16 + l16;       // this lane's hidden column

  const float*    gx  = gx_all + (size_t)dir * T_ * B_ * G_;
  const _Float16* whh = whh16 + (size_t)dir * G_ * H_;
  const float*    h0  = h0_all + (size_t)(2 * LAYER + dir) * B_ * H_;
  const float     bn  = bhhn[dir * H_ + j];

  // --- preload all B fragments of w_hh into registers (12 x v16h = 96 VGPRs) ---
  v16h bfrag[3][4];
#pragma unroll
  for (int c = 0; c < 3; ++c) {
    int n = (wv + 8 * c) * 16 + l16;
#pragma unroll
    for (int kt = 0; kt < 4; ++kt)
      bfrag[c][kt] = *(const v16h*)(whh + (size_t)n * H_ + kt * 32 + hi * 16);
  }

  // --- init hidden state: lane-private f32 copy + f16 LDS copy ---
  float hold[5][8];
#pragma unroll
  for (int mt = 0; mt < 5; ++mt)
#pragma unroll
    for (int r = 0; r < 8; ++r) {
      int b = mt * 16 + r + hi * 8;
      float h = (b < B_) ? h0[b * H_ + j] : 0.0f;
      hold[mt][r] = h;
      if (b < B_) hbuf[b * H_ + j] = (_Float16)h;
    }
  for (int i = tid; i < 2 * 10 * H_; i += 256) {   // zero pad rows 70..79, both bufs
    int buf = i / (10 * H_);
    int rem = i % (10 * H_);
    hbuf[buf * ROWS_ * H_ + (70 + rem / H_) * H_ + (rem % H_)] = (_Float16)0.0f;
  }

  const int t0 = dir ? (T_ - 1) : 0;
  gx_fetch(gx + (size_t)t0 * B_ * G_, gxb, tid);   // prologue: gx[t0] -> buf0
  __syncthreads();

  for (int s = 0; s < T_; ++s) {
    const int t  = dir ? (T_ - 1 - s) : s;
    int tn = dir ? (t - 1) : (t + 1);
    if (tn < 0) tn = 0;
    if (tn > T_ - 1) tn = T_ - 1;
    const int cur = s & 1;
    const int nxt = cur ^ 1;

    // issue next step's gx copy first; latency hides under the GEMM
    gx_fetch(gx + (size_t)tn * B_ * G_, gxb + nxt * (B_ * G_), tid);

    // --- gh = h @ w_hh^T via WMMA: 5 M-tiles x 3 N-tiles x 4 K-steps ---
    v8f acc[5][3] = {};
    const _Float16* hb = hbuf + cur * ROWS_ * H_;
#pragma unroll
    for (int kt = 0; kt < 4; ++kt) {
#pragma unroll
      for (int mt = 0; mt < 5; ++mt) {
        AFrag a;
        const int row = mt * 16 + l16;
        const int k0  = kt * 32 + hi * 16;
        a.q[0] = *(const int4*)(hb + row * H_ + k0);
        a.q[1] = *(const int4*)(hb + row * H_ + k0 + 8);
#pragma unroll
        for (int c = 0; c < 3; ++c)
          acc[mt][c] = wmma_f16(a.v, bfrag[c][kt], acc[mt][c]);
      }
    }

    gx_wait();         // this step's gx batch has landed in LDS
    __syncthreads();   // make all waves' async writes visible

    // --- GRU gate math, fully in-wave; write h to the *other* LDS buffer ---
    const float* gxc = gxb + cur * (B_ * G_);
    _Float16* hnb = hbuf + nxt * ROWS_ * H_;
#pragma unroll
    for (int mt = 0; mt < 5; ++mt) {
#pragma unroll
      for (int r = 0; r < 8; ++r) {
        const int b  = mt * 16 + r + hi * 8;
        const int bs = (b < B_) ? b : (B_ - 1);
        const int base = bs * G_ + j;
        float xr = gxc[base];
        float xz = gxc[base + H_];
        float xn = gxc[base + 2 * H_];
        float rg = sigmoidf_(xr + acc[mt][0][r]);
        float zg = sigmoidf_(xz + acc[mt][1][r]);
        float ng = tanhf_(xn + rg * (acc[mt][2][r] + bn));   // b_hh_n inside r*
        float hn = (1.0f - zg) * ng + zg * hold[mt][r];
        hold[mt][r] = hn;
        if (b < B_) {
          hnb[b * H_ + j] = (_Float16)hn;
          if (LAYER == 0)
            out0h[(size_t)(t * B_ + b) * (2 * H_) + dir * H_ + j] = (_Float16)hn;
          else
            dout[(size_t)(b * T_ + t) * (2 * H_) + dir * H_ + j] = hn;
        }
      }
    }
    __syncthreads();   // order h writes (buf nxt) vs next step's A reads
  }
}

// ---------------- workspace layout (bytes, all 256-aligned) ----------------
static constexpr size_t GXP0_OFF  = 0;                       // 2*T*B*G*4 = 215,040,000
static constexpr size_t GXP1_OFF  = 215040000;               // 215,040,000
static constexpr size_t OUT0H_OFF = 430080000;               // 70000*256*2 = 35,840,000
static constexpr size_t WHH16_OFF = 465920000;               // 4*G*H*2   = 393,216
static constexpr size_t WIH16_OFF = WHH16_OFF + 393216;      // 2*G*256*2 = 393,216
static constexpr size_t BSUM0_OFF = WIH16_OFF + 393216;      // 3,072
static constexpr size_t BSUM1_OFF = BSUM0_OFF + 3072;        // 3,072
static constexpr size_t BHHN_OFF  = BSUM1_OFF + 3072;        // 2,048

extern "C" void kernel_launch(void* const* d_in, const int* in_sizes, int n_in,
                              void* d_out, int out_size, void* d_ws, size_t ws_size,
                              hipStream_t stream) {
  (void)in_sizes; (void)n_in; (void)out_size; (void)ws_size;
  const float* x      = (const float*)d_in[0];
  const float* hidden = (const float*)d_in[1];
  const float* wih0f = (const float*)d_in[2];
  const float* whh0f = (const float*)d_in[3];
  const float* bi0f  = (const float*)d_in[4];
  const float* bh0f  = (const float*)d_in[5];
  const float* wih0b = (const float*)d_in[6];
  const float* whh0b = (const float*)d_in[7];
  const float* bi0b  = (const float*)d_in[8];
  const float* bh0b  = (const float*)d_in[9];
  const float* wih1f = (const float*)d_in[10];
  const float* whh1f = (const float*)d_in[11];
  const float* bi1f  = (const float*)d_in[12];
  const float* bh1f  = (const float*)d_in[13];
  const float* wih1b = (const float*)d_in[14];
  const float* whh1b = (const float*)d_in[15];
  const float* bi1b  = (const float*)d_in[16];
  const float* bh1b  = (const float*)d_in[17];

  char* ws = (char*)d_ws;
  float*     gxp0  = (float*)(ws + GXP0_OFF);
  float*     gxp1  = (float*)(ws + GXP1_OFF);
  _Float16*  out0h = (_Float16*)(ws + OUT0H_OFF);
  _Float16*  whh16 = (_Float16*)(ws + WHH16_OFF);   // [l0f,l0b,l1f,l1b][G][H]
  _Float16*  wih16 = (_Float16*)(ws + WIH16_OFF);   // [f,b][G][2H]
  float*     bsum0 = (float*)(ws + BSUM0_OFF);
  float*     bsum1 = (float*)(ws + BSUM1_OFF);
  float*     bhhn  = (float*)(ws + BHHN_OFF);       // [l0f,l0b,l1f,l1b][H]

  // 1) weight conversion + bias folding
  {
    int total = 4 * G_ * H_ + 2 * G_ * 2 * H_ + 4 * G_ + 4 * H_;
    prep_kernel<<<(total + 255) / 256, 256, 0, stream>>>(
        whh0f, whh0b, whh1f, whh1b, wih1f, wih1b,
        bi0f, bh0f, bi0b, bh0b, bi1f, bh1f, bi1b, bh1b,
        whh16, wih16, bsum0, bsum1, bhhn);
  }
  // 2) layer-0 input gates (K=20 VALU GEMM)
  {
    int total = 2 * T_ * B_ * G_;                       // 53,760,000
    gx0_kernel<<<total / 256, 256, 0, stream>>>(x, wih0f, wih0b, bsum0, gxp0);
  }
  const size_t scan_smem = 2 * ROWS_ * H_ * sizeof(_Float16)  // 40,960
                         + 2 * B_ * G_ * sizeof(float);       // 215,040 -> 256,000 B
  // 3) layer-0 bidirectional scan (writes f16 layer-0 output)
  gru_scan_kernel<0><<<dim3(2), dim3(256), scan_smem, stream>>>(
      gxp0, whh16, hidden, bhhn, out0h, (float*)d_out);
  // 4) layer-1 input gates: WMMA GEMM over all 70000 rows, K=256
  gx1_gemm_kernel<<<dim3(T_ * B_ / 16, 2), dim3(256), 0, stream>>>(
      out0h, wih16, bsum1, gxp1);
  // 5) layer-1 bidirectional scan -> final [B,T,2H] f32 output
  gru_scan_kernel<1><<<dim3(2), dim3(256), scan_smem, stream>>>(
      gxp1, whh16 + 2 * G_ * H_, hidden, bhhn + 2 * H_, out0h, (float*)d_out);
}
